// CubeNet2_78950088835280
// MI455X (gfx1250) — compile-verified, matrix operand order checked by reading, not run on hardware
//
#include <hip/hip_runtime.h>

// Problem constants (match reference)
#define Bsz   32
#define Tsz   2048
#define Csz   256
#define Hsz   512
#define Ssz   16
#define NWG   32          // one WG per 16 h-columns; 32*16 = 512 = H
#define KPAD  800         // [cond 256 | x_prev 16 | pad 16 | h 512]
#define KX0   256         // x_prev slot
#define KH0   288         // h slot
#define TEMPERATURE 0.8f
#define EPS_MIN    (-20.0f)

typedef __attribute__((ext_vector_type(16))) __bf16 bf16x16;
typedef __attribute__((ext_vector_type(8)))  float  f32x8;

// CDNA5 hardware transcendentals: v_tanh_f32 + v_rcp_f32 on the serial path.
#if __has_builtin(__builtin_amdgcn_tanhf)
#define FAST_TANH(x) __builtin_amdgcn_tanhf(x)
#else
#define FAST_TANH(x) (1.f - 2.f * __builtin_amdgcn_rcpf(1.f + __expf(2.f * (x))))
#endif
__device__ __forceinline__ float fast_sigmoid(float x) {
  return __builtin_amdgcn_rcpf(1.f + __expf(-x));
}

// Async global->LDS 16B copy (gfx1250 GLOBAL_LOAD_ASYNC_TO_LDS_B128, ASYNCcnt).
__device__ __forceinline__ void async_copy_b128(unsigned lds_addr, const void* gaddr) {
  asm volatile("global_load_async_to_lds_b128 %0, %1, off"
               :: "v"(lds_addr), "v"(gaddr) : "memory");
}
__device__ __forceinline__ void wait_asynccnt0() {
  asm volatile("s_wait_asynccnt 0" ::: "memory");
}

// Load one 16-bit WMMA A/B fragment from a row-major LDS row.
// Lane (l&15) selects the row (M for A, N for B); hi = l>>4.
// Per-lane data: K = k0+hi*8 .. +7 (VGPR0-3) and K = k0+16+hi*8 .. +7 (VGPR4-7).
__device__ __forceinline__ bf16x16 ld_frag(const __bf16* rowptr, int k0, int hi) {
  bf16x16 r;
  *reinterpret_cast<uint4*>(&r)       = *reinterpret_cast<const uint4*>(rowptr + k0 + hi * 8);
  *(reinterpret_cast<uint4*>(&r) + 1) = *reinterpret_cast<const uint4*>(rowptr + k0 + 16 + hi * 8);
  return r;
}

__global__ void cubenet2_init(unsigned* ctr) {
  if (threadIdx.x == 0 && blockIdx.x == 0) *ctr = 0u;
}

__global__ __launch_bounds__(256, 1)
void cubenet2_lstm_persistent(const float* __restrict__ cond,
                              const float* __restrict__ eps,
                              const float* __restrict__ W_ih,
                              const float* __restrict__ W_hh,
                              const float* __restrict__ b_ih,
                              const float* __restrict__ b_hh,
                              const float* __restrict__ W_out,
                              const float* __restrict__ b_out,
                              float* __restrict__ out,
                              unsigned* __restrict__ barrier_ctr,
                              unsigned short* __restrict__ hbuf_raw)
{
  // Weight-stationary LDS: ~196 KB of the 320 KB/WGP pool.
  __shared__ __bf16 Wlds[64 * KPAD];        // this WG's 64 gate rows, padded-K layout
  __shared__ __bf16 WoutLds[32 * Hsz];      // full W_out (2S x H)
  __shared__ __bf16 Alds[Bsz * KPAD];       // A matrix: [cond | zz | 0 | h] per batch row
  __shared__ float  gatesLds[Bsz * 64];     // this WG's gates tile (i|f|g|o x 16 cols)
  __shared__ float  outLds[Bsz * 32];       // mean|logvar raw
  __shared__ float  cLds[Bsz * 16];         // persistent cell state for this WG's cols

  __bf16* hbuf = reinterpret_cast<__bf16*>(hbuf_raw);  // double buffered [2][B][H]

  const int tid  = threadIdx.x;
  const int w    = blockIdx.x;      // owns h-cols [w*16, w*16+16)
  const int lane = tid & 31;
  const int wid  = tid >> 5;        // 8 waves
  const int hi   = lane >> 4;
  const int ln   = lane & 15;

  // ---- one-time weight preload (fp32 -> bf16), K layout [cond|x|pad|h]
  for (int idx = tid; idx < 64 * KPAD; idx += 256) {
    int r = idx / KPAD;
    int k = idx - r * KPAD;
    int R = (r >> 4) * Hsz + w * 16 + (r & 15);   // global 4H gate-row index
    float v;
    if (k < Csz + Ssz)    v = W_ih[R * (Csz + Ssz) + k];
    else if (k < KH0)     v = 0.f;
    else                  v = W_hh[R * Hsz + (k - KH0)];
    Wlds[idx] = (__bf16)v;
  }
  for (int idx = tid; idx < 32 * Hsz; idx += 256)
    WoutLds[idx] = (__bf16)W_out[idx];
  // zero x_prev + pad + h regions of A (initial state)
  for (int idx = tid; idx < Bsz * (KPAD - KX0); idx += 256) {
    int b = idx / (KPAD - KX0);
    int k = KX0 + (idx - b * (KPAD - KX0));
    Alds[b * KPAD + k] = (__bf16)0.f;
  }
  for (int idx = tid; idx < Bsz * 16; idx += 256) cLds[idx] = 0.f;

  // per-wave tile assignment + fused biases
  const int   mt = wid >> 2;                       // batch-half (gates gemm)
  const int   nt = wid & 3;                        // gate type i,f,g,o
  const int   gcol  = nt * Hsz + w * 16 + ln;
  const float gbias = b_ih[gcol] + b_hh[gcol];
  const int   mo = wid >> 1, no = wid & 1;         // out gemm (waves 0..3)
  const float obias = (wid < 4) ? b_out[no * 16 + ln] : 0.f;

  __syncthreads();

  for (int t = 0; t < Tsz; ++t) {
    // ---- stage cond_t -> A[:, 0:256] (fp32 -> bf16); prefetch next step
    {
      const float* condt = cond + (size_t)t * Csz;
      for (int i = tid; i < Bsz * (Csz / 4); i += 256) {
        int b = i >> 6, c4 = i & 63;
        const float4* p = reinterpret_cast<const float4*>(condt + (size_t)b * Tsz * Csz) + c4;
        float4 v = *p;
        __builtin_prefetch(reinterpret_cast<const float*>(p) + Csz, 0, 1);  // cond[t+1]
        __bf16* d = &Alds[b * KPAD + c4 * 4];
        d[0] = (__bf16)v.x; d[1] = (__bf16)v.y; d[2] = (__bf16)v.z; d[3] = (__bf16)v.w;
      }
    }
    __syncthreads();

    // ---- gates GEMM: 8 waves x one 16x16 f32 tile, K = 800 (25 wmma/wave)
    {
      f32x8 acc;
      #pragma unroll
      for (int g = 0; g < 8; ++g) acc[g] = gbias;
      const __bf16* arow = &Alds[(mt * 16 + ln) * KPAD];
      const __bf16* brow = &Wlds[(nt * 16 + ln) * KPAD];
      #pragma unroll 5
      for (int k0 = 0; k0 < KPAD; k0 += 32) {
        bf16x16 a = ld_frag(arow, k0, hi);
        bf16x16 b = ld_frag(brow, k0, hi);
        acc = __builtin_amdgcn_wmma_f32_16x16x32_bf16(false, a, false, b,
                                                      (short)0, acc, false, false);
      }
      #pragma unroll
      for (int g = 0; g < 8; ++g)
        gatesLds[(mt * 16 + hi * 8 + g) * 64 + nt * 16 + ln] = acc[g];
    }
    __syncthreads();

    // ---- LSTM pointwise update (hardware v_tanh/v_rcp/v_exp on serial path)
    {
      __bf16* hdst = hbuf + (size_t)(t & 1) * Bsz * Hsz;
      for (int idx = tid; idx < Bsz * 16; idx += 256) {
        int b = idx >> 4, j = idx & 15;
        const float* gr = &gatesLds[b * 64];
        float ig = fast_sigmoid(gr[j]);
        float fg = fast_sigmoid(gr[16 + j]);
        float gg = FAST_TANH(gr[32 + j]);
        float og = fast_sigmoid(gr[48 + j]);
        float c  = fg * cLds[idx] + ig * gg;
        cLds[idx] = c;
        float h = og * FAST_TANH(c);
        hdst[b * Hsz + w * 16 + j] = (__bf16)h;
      }
    }

    // ---- device-wide barrier: all 512 h columns visible
    __threadfence();                 // release h stores
    __syncthreads();
    if (tid == 0) {
      __hip_atomic_fetch_add(barrier_ctr, 1u, __ATOMIC_RELEASE, __HIP_MEMORY_SCOPE_AGENT);
      unsigned target = (unsigned)(t + 1) * NWG;
      while (__hip_atomic_load(barrier_ctr, __ATOMIC_ACQUIRE, __HIP_MEMORY_SCOPE_AGENT) < target)
        __builtin_amdgcn_s_sleep(1);
    }
    __syncthreads();
    __threadfence();                 // acquire

    // ---- pull full h_t -> A[:, 288:800] via async direct-to-LDS b128 copies
    {
      const __bf16* hsrc = hbuf + (size_t)(t & 1) * Bsz * Hsz;
      for (int i = tid; i < Bsz * (Hsz / 8); i += 256) {
        int b = i >> 6, j = i & 63;
        unsigned laddr = (unsigned)(uintptr_t)(&Alds[b * KPAD + KH0]) + j * 16u;
        async_copy_b128(laddr, &hsrc[b * Hsz + j * 8]);
      }
      wait_asynccnt0();
    }
    __syncthreads();

    // ---- out GEMM (redundant per WG): 32x32 = h @ W_outT + b_out, K=512
    if (wid < 4) {
      f32x8 acc;
      #pragma unroll
      for (int g = 0; g < 8; ++g) acc[g] = obias;
      const __bf16* arow = &Alds[(mo * 16 + ln) * KPAD];
      const __bf16* brow = &WoutLds[(no * 16 + ln) * Hsz];
      #pragma unroll 4
      for (int k0 = 0; k0 < Hsz; k0 += 32) {
        bf16x16 a = ld_frag(arow, KH0 + k0, hi);
        bf16x16 b = ld_frag(brow, k0, hi);
        acc = __builtin_amdgcn_wmma_f32_16x16x32_bf16(false, a, false, b,
                                                      (short)0, acc, false, false);
      }
      #pragma unroll
      for (int g = 0; g < 8; ++g)
        outLds[(mo * 16 + hi * 8 + g) * 32 + no * 16 + ln] = acc[g];
    }
    __syncthreads();

    // ---- mean/logvar/zz; zz feeds back into A x_prev slot; WG0 writes outputs
    for (int idx = tid; idx < Bsz * Ssz; idx += 256) {
      int b = idx >> 4, s = idx & 15;
      float mean = outLds[b * 32 + s];
      float lv   = fmaxf(outLds[b * 32 + 16 + s], EPS_MIN);
      float e    = eps[((size_t)b * Tsz + t) * Ssz + s];
      float zz   = mean + e * TEMPERATURE * __expf(lv);
      Alds[b * KPAD + KX0 + s] = (__bf16)zz;
      if (w == 0) {
        size_t m = ((size_t)b * Tsz + t) * Ssz + s;
        out[m]                               = mean;
        out[(size_t)Bsz * Tsz * Ssz + m]     = lv;
        out[2 * (size_t)Bsz * Tsz * Ssz + m] = zz;
      }
    }
    __syncthreads();
  }
}

extern "C" void kernel_launch(void* const* d_in, const int* in_sizes, int n_in,
                              void* d_out, int out_size, void* d_ws, size_t ws_size,
                              hipStream_t stream) {
  const float* cond  = (const float*)d_in[0];
  const float* eps   = (const float*)d_in[1];
  const float* W_ih  = (const float*)d_in[2];
  const float* W_hh  = (const float*)d_in[3];
  const float* b_ih  = (const float*)d_in[4];
  const float* b_hh  = (const float*)d_in[5];
  const float* W_out = (const float*)d_in[6];
  const float* b_out = (const float*)d_in[7];
  float* out = (float*)d_out;

  // workspace: [0,16) barrier counter; [16, 16 + 2*B*H*2) double-buffered bf16 h
  unsigned*       ctr  = (unsigned*)d_ws;
  unsigned short* hbuf = (unsigned short*)((char*)d_ws + 16);

  cubenet2_init<<<1, 1, 0, stream>>>(ctr);   // stream-ordered reset (graph-replay safe)
  cubenet2_lstm_persistent<<<NWG, 256, 0, stream>>>(cond, eps, W_ih, W_hh, b_ih, b_hh,
                                                    W_out, b_out, out, ctr, hbuf);
}
